// IG_MSA_36163624633020
// MI455X (gfx1250) — matrix operand order repaired and checked by reading, not compiled, not run on hardware
//
#include <hip/hip_runtime.h>
#include <hip/hip_bf16.h>

#define BATCH  2
#define HGT    256
#define WID    256
#define CDIM   512
#define NHEADS 8
#define DHEAD  64
#define NTOK   (HGT*WID)      // 65536 tokens per batch
#define MTOT   (BATCH*NTOK)   // 131072 total tokens

typedef __attribute__((ext_vector_type(16))) __bf16 v16bf;
typedef __attribute__((ext_vector_type(8)))  __bf16 v8bf;
typedef __attribute__((ext_vector_type(8)))  float  v8f;

__device__ __forceinline__ __bf16 f2bf(float f) {
    union { float f; unsigned u; } a; a.f = f;
    unsigned r = a.u + 0x7FFFu + ((a.u >> 16) & 1u);   // round-to-nearest-even
    union { unsigned short s; __bf16 b; } o; o.s = (unsigned short)(r >> 16);
    return o.b;
}

__device__ __forceinline__ v8f wmma_bf16(v16bf a, v16bf b, v8f c) {
    return __builtin_amdgcn_wmma_f32_16x16x32_bf16(false, a, false, b, (short)0, c, false, false);
}

// A fragment (16x32 bf16) from a [row][k] tile; stride in elements (multiple of 8).
// lanes 0-15: rows M=0..15, K chunks {0..7,16..23}; lanes 16-31: K chunks {8..15,24..31}.
__device__ __forceinline__ v16bf frag_a(const __bf16* base, int stride) {
    int lane = threadIdx.x & 31;
    const __bf16* p = base + (lane & 15) * stride + ((lane >> 4) << 3);
    union { v16bf v; v8bf h[2]; } u;
    u.h[0] = *(const v8bf*)p;          // ds_load_b128 / global_load_b128
    u.h[1] = *(const v8bf*)(p + 16);
    return u.v;
}
// B fragment (32x16 bf16) from a K-transposed [n][k] tile.
// lanes 0-15: N=lane, K chunks {0..7,16..23}; lanes 16-31: N=lane-16, K chunks {8..15,24..31}.
__device__ __forceinline__ v16bf frag_b(const __bf16* base, int stride) {
    int lane = threadIdx.x & 31;
    const __bf16* p = base + (lane & 15) * stride + ((lane >> 4) << 3);
    union { v16bf v; v8bf h[2]; } u;
    u.h[0] = *(const v8bf*)p;
    u.h[1] = *(const v8bf*)(p + 16);
    return u.v;
}

// ---------------------------------------------------------------- small utils
__global__ void zero_buf(float* p, int n) {
    int i = blockIdx.x * 256 + threadIdx.x;
    if (i < n) p[i] = 0.f;
}

__global__ void transpose_wp(const float* __restrict__ Wp, __bf16* __restrict__ Wpt) {
    // Wpt[n][k] = bf16(Wp[k][n]); 512x512
    for (int i = blockIdx.x * blockDim.x + threadIdx.x; i < CDIM * CDIM;
         i += gridDim.x * blockDim.x) {
        int n = i >> 9, k = i & 511;
        Wpt[i] = f2bf(Wp[(size_t)k * CDIM + n]);
    }
}

// --------------------------------------------------- K1: v_inp = x @ Wv (WMMA)
// Block tile 64(M) x 128(N), full K resident in LDS; one barrier per block.
// 8 waves: wave w -> mt = w&3 (M tile), nh = w>>2 (N half, 4 n-tiles each).
__global__ __launch_bounds__(256) void gemm_xw(const float* __restrict__ A,
                                               const float* __restrict__ Bw,
                                               float* __restrict__ C, int N, int K) {
    __shared__ __align__(16) __bf16 As[64 * 520];    // [m][k], k-pad to 520
    __shared__ __align__(16) __bf16 Bs[128 * 520];   // [n][k] transposed
    int tid = threadIdx.x, wid = tid >> 5, lane = tid & 31;
    int bm = blockIdx.x * 64, bn = blockIdx.y * 128;
    int mt = wid & 3, nh = wid >> 2;
    for (int i = tid; i < 64 * 512; i += 256) {
        int r = i >> 9, c = i & 511;
        As[r * 520 + c] = f2bf(A[(size_t)(bm + r) * K + c]);
    }
    for (int i = tid; i < 128 * 512; i += 256) {
        int nn = i & 127, kk = i >> 7;
        Bs[nn * 520 + kk] = f2bf(Bw[(size_t)kk * N + bn + nn]);
    }
    __syncthreads();
    v8f acc[4] = {};
    for (int k0 = 0; k0 < CDIM; k0 += 32) {
        v16bf a = frag_a(As + mt * 16 * 520 + k0, 520);
        #pragma unroll
        for (int j = 0; j < 4; ++j) {
            v16bf b = frag_b(Bs + (nh * 64 + j * 16) * 520 + k0, 520);
            acc[j] = wmma_bf16(a, b, acc[j]);
        }
    }
    int c15 = lane & 15, rb = (lane >> 4) * 8;
    #pragma unroll
    for (int j = 0; j < 4; ++j) {
        int ncol = bn + (nh * 4 + j) * 16 + c15;
        #pragma unroll
        for (int v = 0; v < 8; ++v) {
            size_t row = (size_t)bm + mt * 16 + rb + v;
            C[row * N + ncol] = acc[j][v];
        }
    }
}

// ------------------- K2: fused q/k projection + Gram G = K_raw Q_raw^T + sumsq
// grid (NTOK/2048, NHEADS, BATCH); per-head W resident in LDS; 2 barriers/strip.
__global__ __launch_bounds__(256) void qk_gram(const float* __restrict__ x,
                                               const float* __restrict__ Wq,
                                               const float* __restrict__ Wk,
                                               float* __restrict__ G,
                                               float* __restrict__ sqq,
                                               float* __restrict__ sqk) {
    __shared__ __align__(16) __bf16 wqs[64 * 520];  // Wq_h [n][k] transposed, full K
    __shared__ __align__(16) __bf16 wks[64 * 520];
    __shared__ __align__(16) __bf16 xs[32 * 520];   // x strip [t][k]
    __shared__ __align__(16) __bf16 qs[64 * 40];    // q strip transposed [e][t]
    __shared__ __align__(16) __bf16 ks[64 * 40];    // k strip transposed [d][t]
    int tid = threadIdx.x, wid = tid >> 5, lane = tid & 31;
    int h = blockIdx.y, bi = blockIdx.z;
    size_t tbase = (size_t)bi * NTOK + (size_t)blockIdx.x * 2048;
    int mt = wid & 1, nt = wid >> 1;   // q/k compute: token half, d/e quarter
    int dt = wid & 3, eh = wid >> 2;   // gram: d tile, e half
    int c15 = lane & 15, rb = (lane >> 4) * 8;
    // preload per-head weights, K-transposed (coalesced on n)
    for (int i = tid; i < 64 * 512; i += 256) {
        int nn = i & 63, kk = i >> 6;
        size_t widx = (size_t)kk * CDIM + h * DHEAD + nn;
        wqs[nn * 520 + kk] = f2bf(Wq[widx]);
        wks[nn * 520 + kk] = f2bf(Wk[widx]);
    }
    __syncthreads();
    v8f g0 = {}, g1 = {};
    float sq_q = 0.f, sq_k = 0.f;
    for (int ti = 0; ti < 64; ++ti) {            // 64 strips of 32 tokens
        size_t trow = tbase + (size_t)ti * 32;
        for (int i = tid; i < 32 * 512; i += 256) {
            int r = i >> 9, cc = i & 511;
            xs[r * 520 + cc] = f2bf(x[(trow + r) * CDIM + cc]);
        }
        __syncthreads();                         // barrier A: strip staged
        v8f qa = {}, ka = {};
        for (int k0 = 0; k0 < CDIM; k0 += 32) {  // barrier-free K loop
            v16bf a = frag_a(xs + mt * 16 * 520 + k0, 520);
            qa = wmma_bf16(a, frag_b(wqs + nt * 16 * 520 + k0, 520), qa);
            ka = wmma_bf16(a, frag_b(wks + nt * 16 * 520 + k0, 520), ka);
        }
        // stash strip transposed for the Gram pass; accumulate sum of squares
        int coln = nt * 16 + c15;           // fixed channel per lane
        int t0 = mt * 16 + rb;              // token base (8 consecutive)
        v8bf qp, kp;
        #pragma unroll
        for (int v = 0; v < 8; ++v) {
            float qv = qa[v], kv = ka[v];
            sq_q += qv * qv; sq_k += kv * kv;
            qp[v] = f2bf(qv); kp[v] = f2bf(kv);
        }
        *(v8bf*)(qs + coln * 40 + t0) = qp;
        *(v8bf*)(ks + coln * 40 + t0) = kp;
        __syncthreads();                         // barrier B: q/k strips visible
        // Gram over the 32-token strip: A = k^T [d][t], B = q [t][e]
        v16bf ak = frag_a(ks + dt * 16 * 40, 40);
        g0 = wmma_bf16(ak, frag_b(qs + (eh * 32) * 40, 40), g0);
        g1 = wmma_bf16(ak, frag_b(qs + (eh * 32 + 16) * 40, 40), g1);
    }
    size_t gb = (size_t)(bi * NHEADS + h) * DHEAD * DHEAD;
    #pragma unroll
    for (int v = 0; v < 8; ++v) {
        int d = dt * 16 + rb + v;
        atomicAdd(&G[gb + (size_t)d * DHEAD + eh * 32 + c15],      g0[v]);
        atomicAdd(&G[gb + (size_t)d * DHEAD + eh * 32 + 16 + c15], g1[v]);
    }
    size_t sb2 = (size_t)(bi * NHEADS + h) * DHEAD;
    atomicAdd(&sqq[sb2 + nt * 16 + c15], sq_q);
    atomicAdd(&sqk[sb2 + nt * 16 + c15], sq_k);
}

// ---------------- K3: attn = softmax( G / (|k_d||q_e|) * rescale ), in place
__global__ void softmax_attn(float* __restrict__ G, const float* __restrict__ sqq,
                             const float* __restrict__ sqk,
                             const float* __restrict__ rescale) {
    int bh = blockIdx.x;            // b*NHEADS + h
    int h = bh & (NHEADS - 1);
    int d = threadIdx.x;            // 64 threads, one row each
    float* row = G + (size_t)bh * DHEAD * DHEAD + (size_t)d * DHEAD;
    const float* q2 = sqq + (size_t)bh * DHEAD;
    float nk = fmaxf(sqrtf(sqk[(size_t)bh * DHEAD + d]), 1e-12f);
    float rs = rescale[h];
    float vals[DHEAD];
    float mx = -1e30f;
    for (int e = 0; e < DHEAD; ++e) {
        float nq = fmaxf(sqrtf(q2[e]), 1e-12f);
        float t = row[e] / (nk * nq) * rs;
        vals[e] = t; mx = fmaxf(mx, t);
    }
    float s = 0.f;
    for (int e = 0; e < DHEAD; ++e) { float t = expf(vals[e] - mx); vals[e] = t; s += t; }
    float inv = 1.f / s;
    for (int e = 0; e < DHEAD; ++e) row[e] = vals[e] * inv;
}

// ---- K4: fused out_c = ((v*illu) x blockdiag(attn^T)) @ Wp + bp per 32-token strip
__global__ __launch_bounds__(256) void attn_out(const float* __restrict__ vinp,
                                                const float* __restrict__ illu,
                                                const float* __restrict__ attn,
                                                const __bf16* __restrict__ Wpt,
                                                const float* __restrict__ bp,
                                                float* __restrict__ out) {
    __shared__ __align__(16) __bf16 attnS[NHEADS * DHEAD * DHEAD]; // [h][d][e], stride 64
    __shared__ __align__(16) __bf16 vs[32 * 520];                  // v strip [t][c]
    __shared__ __align__(16) __bf16 tmpS[32 * 520];                // attn-applied strip
    int tid = threadIdx.x, wid = tid >> 5, lane = tid & 31;
    size_t tok0 = (size_t)blockIdx.x * 32;
    int bi = (int)(tok0 >> 16);
    const float* attnB = attn + (size_t)bi * NHEADS * DHEAD * DHEAD;
    for (int i = tid; i < NHEADS * DHEAD * DHEAD; i += 256) attnS[i] = f2bf(attnB[i]);
    for (int i = tid; i < 32 * CDIM; i += 256) {
        int t = i >> 9, c = i & 511;
        size_t gi = (tok0 + t) * CDIM + c;
        vs[t * 520 + c] = f2bf(vinp[gi] * illu[gi]);
    }
    __syncthreads();
    int mt = wid & 1, ng = wid >> 1;
    int c15 = lane & 15, rb = (lane >> 4) * 8;
    // GEMM1: tmp[t][h*64+d] = sum_e v[t][h*64+e] * attn[h][d][e]  (K=64 per head)
    for (int i = 0; i < 8; ++i) {
        int ntile = ng * 8 + i;
        int h = ntile >> 2, dt = ntile & 3;
        v8f acc = {};
        #pragma unroll
        for (int j = 0; j < 2; ++j) {
            v16bf a = frag_a(vs + mt * 16 * 520 + h * 64 + j * 32, 520);
            v16bf b = frag_b(attnS + (h * 64 + dt * 16) * 64 + j * 32, 64);
            acc = wmma_bf16(a, b, acc);
        }
        int col = h * 64 + dt * 16 + c15;
        int r0 = mt * 16 + rb;
        #pragma unroll
        for (int v = 0; v < 8; ++v) tmpS[(r0 + v) * 520 + col] = f2bf(acc[v]);
    }
    __syncthreads();
    // GEMM2: out[t][n] = sum_k tmp[t][k] * Wp[k][n] + bp[n]; B frags direct from
    // bf16-transposed Wp in global (L2 resident), two 16B loads per fragment.
    for (int i = 0; i < 8; ++i) {
        int n0 = (ng * 8 + i) * 16;
        v8f acc = {};
        for (int k0 = 0; k0 < CDIM; k0 += 32) {
            v16bf a = frag_a(tmpS + mt * 16 * 520 + k0, 520);
            v16bf b = frag_b(Wpt + (size_t)n0 * CDIM + k0, CDIM);
            acc = wmma_bf16(a, b, acc);
        }
        float bias = bp[n0 + c15];
        int r0 = mt * 16 + rb;
        #pragma unroll
        for (int v = 0; v < 8; ++v)
            out[(tok0 + r0 + v) * CDIM + n0 + c15] = acc[v] + bias;
    }
}

// ------------- K5/K6: depthwise 3x3 (SAME, cross-correlation), opt GELU / accum
// Halo tile staged with CDNA5 async LDS copies (ASYNCcnt), OOB chunks zeroed.
__global__ __launch_bounds__(256) void dwconv3x3_k(const float* __restrict__ src,
                                                   const float* __restrict__ kern,
                                                   float* __restrict__ dst,
                                                   int do_gelu, int accum) {
    __shared__ __align__(16) float tile[10 * 10 * 64];  // 8x8 pixels + halo, 64 ch
    int cg = blockIdx.z & 7;                 // CDIM/64 = 8 channel groups
    int bi = blockIdx.z >> 3;
    int x0 = blockIdx.x * 8, y0 = blockIdx.y * 8;
    int c0 = cg * 64;
    const float* sb = src + (size_t)bi * NTOK * CDIM;
    // 1600 16-byte chunks: chunk = (pixel, 4-channel group)
    for (int ch = threadIdx.x; ch < 1600; ch += 256) {
        int c4 = ch & 15, pix = ch >> 4;
        int gx = x0 + (pix % 10) - 1, gy = y0 + (pix / 10) - 1;
        float* lp = tile + ch * 4;
        if (gx >= 0 && gx < WID && gy >= 0 && gy < HGT) {
            const float* gp = sb + ((size_t)gy * WID + gx) * CDIM + c0 + c4 * 4;
            unsigned loff = (unsigned)(size_t)lp;     // LDS byte offset (low 32 bits)
            asm volatile("global_load_async_to_lds_b128 %0, %1, off"
                         :: "v"(loff), "v"(gp) : "memory");
        } else {
            lp[0] = 0.f; lp[1] = 0.f; lp[2] = 0.f; lp[3] = 0.f;
        }
    }
    asm volatile("s_wait_asynccnt 0x0" ::: "memory");
    __syncthreads();
    int c = threadIdx.x & 63, pg = threadIdx.x >> 6;
    float w9[9];
    #pragma unroll
    for (int j = 0; j < 9; ++j) w9[j] = kern[(size_t)(c0 + c) * 9 + j];
    for (int pp = 0; pp < 16; ++pp) {
        int p = pg * 16 + pp, px = p & 7, py = p >> 3;
        float s = 0.f;
        #pragma unroll
        for (int ky = 0; ky < 3; ++ky)
            #pragma unroll
            for (int kx = 0; kx < 3; ++kx)
                s += w9[ky * 3 + kx] * tile[((py + ky) * 10 + px + kx) * 64 + c];
        if (do_gelu) s = 0.5f * s * (1.f + erff(s * 0.70710678f));  // exact GELU
        size_t oi = ((size_t)bi * NTOK + (size_t)(y0 + py) * WID + (x0 + px)) * CDIM + c0 + c;
        if (accum) dst[oi] += s; else dst[oi] = s;
    }
}

extern "C" void kernel_launch(void* const* d_in, const int* in_sizes, int n_in,
                              void* d_out, int out_size, void* d_ws, size_t ws_size,
                              hipStream_t stream) {
    (void)in_sizes; (void)n_in; (void)out_size; (void)ws_size;
    const float* x    = (const float*)d_in[0];
    const float* illu = (const float*)d_in[1];
    const float* Wq   = (const float*)d_in[2];
    const float* Wk   = (const float*)d_in[3];
    const float* Wv   = (const float*)d_in[4];
    const float* resc = (const float*)d_in[5];
    const float* Wp   = (const float*)d_in[6];
    const float* bp   = (const float*)d_in[7];
    const float* cv1  = (const float*)d_in[8];
    const float* cv2  = (const float*)d_in[9];
    float* out = (float*)d_out;

    char* ws = (char*)d_ws;
    float* v_inp = (float*)ws;                                   // 268 MB
    float* pos1  = (float*)(ws + (size_t)MTOT * CDIM * 4);       // 268 MB
    float* G     = (float*)(ws + (size_t)MTOT * CDIM * 8);       // 256 KB
    float* sqq   = G + BATCH * NHEADS * DHEAD * DHEAD;           // 4 KB
    float* sqk   = sqq + BATCH * NHEADS * DHEAD;                 // 4 KB
    __bf16* Wpt  = (__bf16*)(sqk + BATCH * NHEADS * DHEAD);      // 512 KB

    int nacc = BATCH * NHEADS * DHEAD * DHEAD + 2 * BATCH * NHEADS * DHEAD;
    zero_buf<<<(nacc + 255) / 256, 256, 0, stream>>>(G, nacc);
    transpose_wp<<<256, 256, 0, stream>>>(Wp, Wpt);
    gemm_xw<<<dim3(MTOT / 64, CDIM / 128), 256, 0, stream>>>(x, Wv, v_inp, CDIM, CDIM);
    qk_gram<<<dim3(NTOK / 2048, NHEADS, BATCH), 256, 0, stream>>>(x, Wq, Wk, G, sqq, sqk);
    softmax_attn<<<BATCH * NHEADS, 64, 0, stream>>>(G, sqq, sqk, resc);
    attn_out<<<MTOT / 32, 256, 0, stream>>>(v_inp, illu, G, Wpt, bp, out);
    dwconv3x3_k<<<dim3(WID / 8, HGT / 8, BATCH * (CDIM / 64)), 256, 0, stream>>>(
        v_inp, cv1, pos1, 1, 0);
    dwconv3x3_k<<<dim3(WID / 8, HGT / 8, BATCH * (CDIM / 64)), 256, 0, stream>>>(
        pos1, cv2, out, 0, 1);
}
